// LinearJointTSAttention_71287867179225
// MI455X (gfx1250) — compile-verified, matrix operand order checked by reading, not run on hardware
//
#include <hip/hip_runtime.h>

// ---------------------------------------------------------------------------
// Types
// ---------------------------------------------------------------------------
typedef __attribute__((ext_vector_type(16))) __bf16    v16bf;
typedef __attribute__((ext_vector_type(8)))  float     v8f;
typedef __attribute__((ext_vector_type(4)))  unsigned  v4u;
typedef __attribute__((ext_vector_type(4)))  float     v4f;

union Frag16 {
    v16bf v;
    v4u   q[2];
};
union Pack16 {
    v4u            q[2];
    unsigned short s[16];
};

__device__ __forceinline__ float bf16_to_f32(unsigned short h) {
    unsigned u = ((unsigned)h) << 16;
    return __builtin_bit_cast(float, u);
}
__device__ __forceinline__ unsigned short f32_to_bf16(float f) {
    unsigned u = __builtin_bit_cast(unsigned, f);
    u += 0x7fffu + ((u >> 16) & 1u);   // round-to-nearest-even
    return (unsigned short)(u >> 16);
}
__device__ __forceinline__ v8f zero8() {
    v8f z = {0.f, 0.f, 0.f, 0.f, 0.f, 0.f, 0.f, 0.f};
    return z;
}
__device__ __forceinline__ v8f wmma_bf16(const Frag16& a, const Frag16& b, v8f c) {
    return __builtin_amdgcn_wmma_f32_16x16x32_bf16(
        /*neg_a=*/false, a.v, /*neg_b=*/false, b.v,
        /*c_mod=*/(short)0, c, /*reuse_a=*/false, /*reuse_b=*/false);
}

// Async 32-byte copy global->LDS (two B128 ops; INST_OFFSET applies to both
// the LDS destination and the global source address per CDNA5 ISA 10.x).
__device__ __forceinline__ void async_copy32(unsigned lds_byte_off,
                                             const unsigned short* gptr) {
    asm volatile("global_load_async_to_lds_b128 %0, %1, off"
                 :: "v"(lds_byte_off), "v"(gptr) : "memory");
    asm volatile("global_load_async_to_lds_b128 %0, %1, off offset:16"
                 :: "v"(lds_byte_off), "v"(gptr) : "memory");
}
__device__ __forceinline__ unsigned lds_off(const void* p) {
    // Generic pointers to LDS carry the LDS byte offset in the low 32 bits.
    return (unsigned)(unsigned long long)p;
}

// Problem constants (from setup_inputs)
#define CC    1024
#define LL    3136            // f * N = 16 * 196
#define MROWS 25088           // B * N
#define QKVN  3072
#define SCALE 0.125f          // 64^{-0.5}
#define EPS   1e-6f

// ---------------------------------------------------------------------------
// Kernel 1: fp32 -> bf16 conversion (8 elements / thread)
// ---------------------------------------------------------------------------
__global__ __launch_bounds__(256) void cvt_bf16_kernel(const float* __restrict__ in,
                                                       unsigned short* __restrict__ out,
                                                       long long n8) {
    long long i = (long long)blockIdx.x * blockDim.x + threadIdx.x;
    if (i >= n8) return;
    const v4f* p = (const v4f*)(in + i * 8);
    v4f a = p[0], b = p[1];
    v4u o;
    o.x = (unsigned)f32_to_bf16(a.x) | ((unsigned)f32_to_bf16(a.y) << 16);
    o.y = (unsigned)f32_to_bf16(a.z) | ((unsigned)f32_to_bf16(a.w) << 16);
    o.z = (unsigned)f32_to_bf16(b.x) | ((unsigned)f32_to_bf16(b.y) << 16);
    o.w = (unsigned)f32_to_bf16(b.z) | ((unsigned)f32_to_bf16(b.w) << 16);
    *(v4u*)(out + i * 8) = o;
}

// ---------------------------------------------------------------------------
// Kernel 2/5: NT GEMM  D[M,N] = A[M,K] * B[N,K]^T  (bf16 in, f32 accum)
// Block tile 128x128, BK=32, 256 threads (8 waves), wave tile 32x64 (2x4 frags)
// Double-buffered LDS fed by GLOBAL_LOAD_ASYNC_TO_LDS_B128 (ASYNCcnt).
// ---------------------------------------------------------------------------
template <int OUT_BF16>
__global__ __launch_bounds__(256) void gemm_nt_kernel(const unsigned short* __restrict__ A,
                                                      const unsigned short* __restrict__ B,
                                                      const float* __restrict__ bias,
                                                      float* __restrict__ Df,
                                                      unsigned short* __restrict__ Dh,
                                                      int M, int N, int K) {
    __shared__ unsigned short As[2][128 * 40];
    __shared__ unsigned short Bs[2][128 * 40];

    const int bm   = blockIdx.y * 128;
    const int bn   = blockIdx.x * 128;
    const int tid  = threadIdx.x;
    const int lane = tid & 31;
    const int w    = tid >> 5;
    const int mw   = (w >> 1) * 32;      // wave row offset in tile
    const int nw   = (w & 1) * 64;       // wave col offset in tile
    const int lr   = lane & 15;          // row within 16x16 frag
    const int k0   = (lane >> 4) * 8;    // per-lane K chunk base (ISA layout)
    const int hi   = (lane >> 4) * 8;    // C/D row offset for upper lanes

    // staging: thread -> (row = tid>>1, 16 cols starting at (tid&1)*16)
    const int srow = tid >> 1;
    const int scol = (tid & 1) * 16;

    const unsigned short* agBase = A + (size_t)(bm + srow) * K + scol;
    const unsigned short* bgBase = B + (size_t)(bn + srow) * K + scol;
    const unsigned ldsA[2] = {lds_off(&As[0][srow * 40 + scol]),
                              lds_off(&As[1][srow * 40 + scol])};
    const unsigned ldsB[2] = {lds_off(&Bs[0][srow * 40 + scol]),
                              lds_off(&Bs[1][srow * 40 + scol])};

    v8f acc[2][4];
#pragma unroll
    for (int i = 0; i < 2; ++i)
#pragma unroll
        for (int j = 0; j < 4; ++j) acc[i][j] = zero8();

    // prologue: stage tile 0 into buffer 0 (4 async B128 ops / thread)
    async_copy32(ldsA[0], agBase);
    async_copy32(ldsB[0], bgBase);

    int buf = 0;
    for (int kk = 0; kk < K; kk += 32) {
        if (kk + 32 < K) {
            // prefetch next tile into the other buffer, then wait for the
            // OLDER 4 async ops only (async loads complete in order).
            async_copy32(ldsA[buf ^ 1], agBase + kk + 32);
            async_copy32(ldsB[buf ^ 1], bgBase + kk + 32);
            asm volatile("s_wait_asynccnt 0x4" ::: "memory");
        } else {
            asm volatile("s_wait_asynccnt 0x0" ::: "memory");
        }
        __syncthreads();   // current buffer fully resident for all waves

        Frag16 af[2], bf_[4];
#pragma unroll
        for (int i = 0; i < 2; ++i) {
            const int r = mw + i * 16 + lr;
            af[i].q[0] = *(const v4u*)(&As[buf][r * 40 + k0]);
            af[i].q[1] = *(const v4u*)(&As[buf][r * 40 + k0 + 16]);
        }
#pragma unroll
        for (int j = 0; j < 4; ++j) {
            const int r = nw + j * 16 + lr;
            bf_[j].q[0] = *(const v4u*)(&Bs[buf][r * 40 + k0]);
            bf_[j].q[1] = *(const v4u*)(&Bs[buf][r * 40 + k0 + 16]);
        }
#pragma unroll
        for (int i = 0; i < 2; ++i)
#pragma unroll
            for (int j = 0; j < 4; ++j)
                acc[i][j] = wmma_bf16(af[i], bf_[j], acc[i][j]);

        __syncthreads();   // all reads of this buffer done before it is refilled
        buf ^= 1;
    }

    // epilogue: C/D layout: VGPR r -> row (r + hi), col = lane&15 within tile
#pragma unroll
    for (int i = 0; i < 2; ++i) {
#pragma unroll
        for (int j = 0; j < 4; ++j) {
            const int col = bn + nw + j * 16 + lr;
#pragma unroll
            for (int r = 0; r < 8; ++r) {
                const int row = bm + mw + i * 16 + hi + r;
                const float v = acc[i][j][r];
                if (OUT_BF16) {
                    Dh[(size_t)row * N + col] = f32_to_bf16(v);
                } else {
                    Df[(size_t)row * N + col] = v + bias[col];
                }
            }
        }
    }
}

// ---------------------------------------------------------------------------
// Kernel 3: per-(b,h)  kv[e1,e2] = sum_n relu_scale(k)[n,e1] * v[n,e2]
//           plus k_sum[e] = sum_n relu_scale(k)[n,e]
// One block per (b,h); K-loop over L=3136 in chunks of 32.
// ---------------------------------------------------------------------------
__global__ __launch_bounds__(256) void kv_kernel(const unsigned short* __restrict__ qkv,
                                                 float* __restrict__ kv,
                                                 float* __restrict__ ksum) {
    __shared__ unsigned short kT[64 * 40];   // [e1][n]  (transposed, K-contiguous)
    __shared__ unsigned short vT[64 * 40];   // [e2][n]

    const int bh  = blockIdx.x;          // 0..127
    const int bb  = bh >> 4;
    const int h   = bh & 15;
    const int tid = threadIdx.x;
    const int lane = tid & 31;
    const int w    = tid >> 5;
    const int lr   = lane & 15;
    const int k0   = (lane >> 4) * 8;
    const int hi   = (lane >> 4) * 8;

    const int e1t  = (w >> 1) * 16;      // wave's e1 tile (4 tiles)
    const int e2t0 = (w & 1) * 32;       // wave's e2 tile pair (2 tiles)

    v8f acc[2];
    acc[0] = zero8();
    acc[1] = zero8();
    float ks = 0.f;

    // staging: tid>>7 selects k(0)/v(1); local: n = (tid&127)>>2, e0 = (tid&3)*16
    const int mtx = tid >> 7;
    const int loc = tid & 127;
    const int sn  = loc >> 2;
    const int se  = (loc & 3) * 16;
    const unsigned short* src =
        qkv + (size_t)bb * LL * QKVN + (mtx ? 2048 : 1024) + h * 64 + se;

    for (int n0 = 0; n0 < LL; n0 += 32) {
        const unsigned short* p = src + (size_t)(n0 + sn) * QKVN;
        Pack16 d;
        d.q[0] = *(const v4u*)p;
        d.q[1] = *(const v4u*)(p + 8);

        __syncthreads();
        if (mtx == 0) {
#pragma unroll
            for (int i = 0; i < 16; ++i) {
                float f = bf16_to_f32(d.s[i]);
                f = fmaxf(f, 0.f) + SCALE;                    // relu(k)+scale
                kT[(se + i) * 40 + sn] = f32_to_bf16(f);
            }
        } else {
#pragma unroll
            for (int i = 0; i < 16; ++i) vT[(se + i) * 40 + sn] = d.s[i];
        }
        __syncthreads();

        if (tid < 64) {
#pragma unroll 8
            for (int nn = 0; nn < 32; ++nn) ks += bf16_to_f32(kT[tid * 40 + nn]);
        }

        Frag16 a;
        a.q[0] = *(const v4u*)(&kT[(e1t + lr) * 40 + k0]);
        a.q[1] = *(const v4u*)(&kT[(e1t + lr) * 40 + k0 + 16]);
#pragma unroll
        for (int j = 0; j < 2; ++j) {
            Frag16 b;
            b.q[0] = *(const v4u*)(&vT[(e2t0 + j * 16 + lr) * 40 + k0]);
            b.q[1] = *(const v4u*)(&vT[(e2t0 + j * 16 + lr) * 40 + k0 + 16]);
            acc[j] = wmma_bf16(a, b, acc[j]);
        }
    }

    float* kvp = kv + (size_t)bh * 64 * 64;
#pragma unroll
    for (int j = 0; j < 2; ++j)
#pragma unroll
        for (int r = 0; r < 8; ++r)
            kvp[(e1t + hi + r) * 64 + e2t0 + j * 16 + lr] = acc[j][r];

    if (tid < 64) ksum[(size_t)bh * 64 + tid] = ks;
}

// ---------------------------------------------------------------------------
// Kernel 4: out[n,e2] = z[n] * sum_e1 q_[n,e1] * kv[e1,e2]
//           z[n] = 1 / (q_[n,:] . k_sum + eps)
// Block: 64 rows of one (b,h); grid (49, 128)
// ---------------------------------------------------------------------------
__global__ __launch_bounds__(256) void out_kernel(const unsigned short* __restrict__ qkv,
                                                  const float* __restrict__ kv,
                                                  const float* __restrict__ ksum,
                                                  unsigned short* __restrict__ outb) {
    __shared__ unsigned short qs[64 * 72];    // [m][e1]  K-contiguous for A
    __shared__ unsigned short kvT[64 * 72];   // [e2][e1] K-contiguous for B
    __shared__ float zsh[64];
    __shared__ float kss[64];

    const int mt  = blockIdx.x;               // 0..48  (3136 / 64)
    const int bh  = blockIdx.y;               // 0..127
    const int bb  = bh >> 4;
    const int h   = bh & 15;
    const int tid = threadIdx.x;
    const int lane = tid & 31;
    const int w    = tid >> 5;
    const int lr   = lane & 15;
    const int k0   = (lane >> 4) * 8;
    const int hi   = (lane >> 4) * 8;
    const int m0   = mt * 64;

    // stage q (relu+scale fused): row = tid>>2, 16 cols at (tid&3)*16
    const int srow = tid >> 2;
    const int se   = (tid & 3) * 16;
    {
        const unsigned short* qp =
            qkv + ((size_t)bb * LL + m0 + srow) * QKVN + h * 64 + se;
        Pack16 d;
        d.q[0] = *(const v4u*)qp;
        d.q[1] = *(const v4u*)(qp + 8);
#pragma unroll
        for (int i = 0; i < 16; ++i) {
            float f = bf16_to_f32(d.s[i]);
            qs[srow * 72 + se + i] = f32_to_bf16(fmaxf(f, 0.f) + SCALE);
        }
    }
    // stage kv transposed (fp32 -> bf16)
    {
        const float* kvp = kv + (size_t)bh * 4096;
#pragma unroll
        for (int i = 0; i < 16; ++i) {
            const int e1 = srow, e2 = se + i;
            kvT[e2 * 72 + e1] = f32_to_bf16(kvp[e1 * 64 + e2]);
        }
    }
    if (tid < 64) kss[tid] = ksum[(size_t)bh * 64 + tid];
    __syncthreads();

    if (tid < 64) {
        float zz = 0.f;
#pragma unroll 8
        for (int e = 0; e < 64; ++e) zz += bf16_to_f32(qs[tid * 72 + e]) * kss[e];
        zsh[tid] = 1.0f / (zz + EPS);
    }

    // wave tiles: m = (w>>1)*16, cols = (w&1)*32 + {0,16}
    const int mwt = (w >> 1) * 16;
    const int nwt = (w & 1) * 32;
    v8f acc[2];
    acc[0] = zero8();
    acc[1] = zero8();
#pragma unroll
    for (int kk = 0; kk < 64; kk += 32) {
        Frag16 a;
        a.q[0] = *(const v4u*)(&qs[(mwt + lr) * 72 + kk + k0]);
        a.q[1] = *(const v4u*)(&qs[(mwt + lr) * 72 + kk + k0 + 16]);
#pragma unroll
        for (int j = 0; j < 2; ++j) {
            Frag16 b;
            b.q[0] = *(const v4u*)(&kvT[(nwt + j * 16 + lr) * 72 + kk + k0]);
            b.q[1] = *(const v4u*)(&kvT[(nwt + j * 16 + lr) * 72 + kk + k0 + 16]);
            acc[j] = wmma_bf16(a, b, acc[j]);
        }
    }
    __syncthreads();   // zsh visible to all

#pragma unroll
    for (int j = 0; j < 2; ++j) {
        const int col = h * 64 + nwt + j * 16 + lr;
#pragma unroll
        for (int r = 0; r < 8; ++r) {
            const int ml = mwt + hi + r;
            const float v = acc[j][r] * zsh[ml];
            outb[((size_t)bb * LL + m0 + ml) * CC + col] = f32_to_bf16(v);
        }
    }
}

// ---------------------------------------------------------------------------
// Host launcher
// ---------------------------------------------------------------------------
extern "C" void kernel_launch(void* const* d_in, const int* in_sizes, int n_in,
                              void* d_out, int out_size, void* d_ws, size_t ws_size,
                              hipStream_t stream) {
    const float* x      = (const float*)d_in[0];   // 128*196*1024
    const float* w_qkv  = (const float*)d_in[1];   // 3072*1024
    const float* w_proj = (const float*)d_in[2];   // 1024*1024
    const float* b_proj = (const float*)d_in[3];   // 1024
    (void)in_sizes; (void)n_in; (void)out_size; (void)ws_size;

    const size_t NX    = (size_t)MROWS * CC;          // 25,690,112
    const size_t NWQKV = (size_t)QKVN * CC;           //  3,145,728
    const size_t NWPRJ = (size_t)CC * CC;             //  1,048,576
    const size_t NQKV  = (size_t)MROWS * QKVN;        // 77,070,336

    unsigned short* xb     = (unsigned short*)d_ws;
    unsigned short* wqkvb  = xb + NX;
    unsigned short* wprojb = wqkvb + NWQKV;
    unsigned short* qkvb   = wprojb + NWPRJ;
    float*          kvb    = (float*)(qkvb + NQKV);
    float*          ksumb  = kvb + (size_t)128 * 64 * 64;
    unsigned short* outb   = (unsigned short*)(ksumb + 128 * 64);

    // 1. fp32 -> bf16 conversions
    cvt_bf16_kernel<<<(unsigned)((NX / 8 + 255) / 256), 256, 0, stream>>>(x, xb, NX / 8);
    cvt_bf16_kernel<<<(unsigned)((NWQKV / 8 + 255) / 256), 256, 0, stream>>>(w_qkv, wqkvb, NWQKV / 8);
    cvt_bf16_kernel<<<(unsigned)((NWPRJ / 8 + 255) / 256), 256, 0, stream>>>(w_proj, wprojb, NWPRJ / 8);

    // 2. qkv = x @ w_qkv^T  (25088 x 3072 x 1024), bf16 out
    gemm_nt_kernel<1><<<dim3(QKVN / 128, MROWS / 128), 256, 0, stream>>>(
        xb, wqkvb, nullptr, nullptr, qkvb, MROWS, QKVN, CC);

    // 3. kv + k_sum per (b,h)
    kv_kernel<<<128, 256, 0, stream>>>(qkvb, kvb, ksumb);

    // 4. out = z * (q_ @ kv), bf16 out
    out_kernel<<<dim3(LL / 64, 128), 256, 0, stream>>>(qkvb, kvb, ksumb, outb);

    // 5. final = out @ w_proj^T + b_proj  (25088 x 1024 x 1024), fp32 out
    gemm_nt_kernel<0><<<dim3(CC / 128, MROWS / 128), 256, 0, stream>>>(
        outb, wprojb, b_proj, (float*)d_out, nullptr, MROWS, CC, CC);
}
